// InfiniAttention_89945205113178
// MI455X (gfx1250) — compile-verified, matrix operand order checked by reading, not run on hardware
//
#include <hip/hip_runtime.h>
#include <hip/hip_bf16.h>

typedef __bf16 bf16;
typedef __attribute__((ext_vector_type(16))) __bf16 v16bf;
typedef __attribute__((ext_vector_type(8)))  float  v8f;

#define WMMA_BF16(a, b, c) \
  __builtin_amdgcn_wmma_f32_16x16x32_bf16(false, (a), false, (b), (short)0, (c), false, false)

static constexpr int E_    = 1024;
static constexpr int H_    = 16;
static constexpr int MEM_  = 256;
static constexpr int B_    = 2;
static constexpr int S_    = 2048;
static constexpr int ROWS_ = B_ * S_;       // 4096
static constexpr float SCALE_ = 0.125f;     // D^-0.5, D=64

// ---------------------------------------------------------------------------
// fp32 -> bf16 conversion
// ---------------------------------------------------------------------------
__global__ void cvt_f32_bf16(const float* __restrict__ in, bf16* __restrict__ out, int n) {
  int i = blockIdx.x * blockDim.x + threadIdx.x;
  if (i < n) out[i] = (bf16)in[i];
}

// memory [256,1024] fp32 -> memoryT [1024,256] bf16
__global__ void cvt_transpose_mem(const float* __restrict__ in, bf16* __restrict__ out) {
  int i = blockIdx.x * blockDim.x + threadIdx.x;   // 0 .. 256*1024-1
  int m = i >> 10;
  int e = i & 1023;
  out[(size_t)e * MEM_ + m] = (bf16)in[i];
}

// ---------------------------------------------------------------------------
// WMMA GEMM: out[M,N] = accScale * (A[M,K] @ Bt[N,K]^T) + bias + addend
// Block = 128 threads (4 waves). Wave owns a 32x64 C tile (8 accumulators);
// WG tile = 128x64. K-loop unrolled x2 with disjoint fragment buffers so the
// global_load clauses for step i+1 issue before the WMMAs of step i
// (avoids s_wait_loadcnt 0 in front of every v_wmma).
// ---------------------------------------------------------------------------
__launch_bounds__(128)
__global__ void gemm_bf16_wmma(const bf16* __restrict__ A, const bf16* __restrict__ Bt,
                               const float* __restrict__ bias, const bf16* __restrict__ addend,
                               float accScale, void* __restrict__ out, int outBf16,
                               int Ndim, int Kdim) {
  const int tid  = threadIdx.x;
  const int wave = tid >> 5;
  const int lane = tid & 31;
  const int lo   = lane & 15;
  const int hi   = lane >> 4;
  const int mBase = blockIdx.y * 128 + wave * 32;
  const int nBase = blockIdx.x * 64;

  // Per-lane fragment base pointers (advance by k inside the loop)
  const bf16* a0p = A + (size_t)(mBase + lo) * Kdim + hi * 16;
  const bf16* a1p = A + (size_t)(mBase + 16 + lo) * Kdim + hi * 16;
  const bf16* bp0 = Bt + (size_t)(nBase + 0 * 16 + lo) * Kdim + hi * 16;
  const bf16* bp1 = Bt + (size_t)(nBase + 1 * 16 + lo) * Kdim + hi * 16;
  const bf16* bp2 = Bt + (size_t)(nBase + 2 * 16 + lo) * Kdim + hi * 16;
  const bf16* bp3 = Bt + (size_t)(nBase + 3 * 16 + lo) * Kdim + hi * 16;

  v8f acc[2][4] = {};

  // Buffer 0: fragments for k-step "kk"
  v16bf aA[2], bA[4];
  // Buffer 1: fragments for k-step "kk+32"
  v16bf aB[2], bB[4];

  aA[0] = *reinterpret_cast<const v16bf*>(a0p);
  aA[1] = *reinterpret_cast<const v16bf*>(a1p);
  bA[0] = *reinterpret_cast<const v16bf*>(bp0);
  bA[1] = *reinterpret_cast<const v16bf*>(bp1);
  bA[2] = *reinterpret_cast<const v16bf*>(bp2);
  bA[3] = *reinterpret_cast<const v16bf*>(bp3);

  for (int kk = 0; kk < Kdim; kk += 64) {
    // Prefetch step kk+32 into buffer B before consuming buffer A
    aB[0] = *reinterpret_cast<const v16bf*>(a0p + kk + 32);
    aB[1] = *reinterpret_cast<const v16bf*>(a1p + kk + 32);
    bB[0] = *reinterpret_cast<const v16bf*>(bp0 + kk + 32);
    bB[1] = *reinterpret_cast<const v16bf*>(bp1 + kk + 32);
    bB[2] = *reinterpret_cast<const v16bf*>(bp2 + kk + 32);
    bB[3] = *reinterpret_cast<const v16bf*>(bp3 + kk + 32);

#pragma unroll
    for (int ns = 0; ns < 4; ++ns) acc[0][ns] = WMMA_BF16(aA[0], bA[ns], acc[0][ns]);
#pragma unroll
    for (int ns = 0; ns < 4; ++ns) acc[1][ns] = WMMA_BF16(aA[1], bA[ns], acc[1][ns]);

    // Prefetch step kk+64 into buffer A (skipped on last iteration)
    if (kk + 64 < Kdim) {
      aA[0] = *reinterpret_cast<const v16bf*>(a0p + kk + 64);
      aA[1] = *reinterpret_cast<const v16bf*>(a1p + kk + 64);
      bA[0] = *reinterpret_cast<const v16bf*>(bp0 + kk + 64);
      bA[1] = *reinterpret_cast<const v16bf*>(bp1 + kk + 64);
      bA[2] = *reinterpret_cast<const v16bf*>(bp2 + kk + 64);
      bA[3] = *reinterpret_cast<const v16bf*>(bp3 + kk + 64);
    }

#pragma unroll
    for (int ns = 0; ns < 4; ++ns) acc[0][ns] = WMMA_BF16(aB[0], bB[ns], acc[0][ns]);
#pragma unroll
    for (int ns = 0; ns < 4; ++ns) acc[1][ns] = WMMA_BF16(aB[1], bB[ns], acc[1][ns]);
  }

#pragma unroll
  for (int mi = 0; mi < 2; ++mi) {
#pragma unroll
    for (int ns = 0; ns < 4; ++ns) {
#pragma unroll
      for (int r = 0; r < 8; ++r) {
        const int row = mBase + mi * 16 + r + 8 * hi;
        const int col = nBase + ns * 16 + lo;
        float v = acc[mi][ns][r] * accScale;
        if (bias)   v += bias[col];
        if (addend) v += (float)addend[(size_t)row * Ndim + col];
        if (outBf16) ((bf16*)out)[(size_t)row * Ndim + col]  = (bf16)v;
        else         ((float*)out)[(size_t)row * Ndim + col] = v;
      }
    }
  }
}

// ---------------------------------------------------------------------------
// Flash attention: one WG per (b, h, 64 q-rows); 4 waves x 16 q-rows each.
// Q/K frags loaded directly from global (contiguous 32B per lane, L2-served).
// P converted C-layout->A-layout via LDS; V tile staged transposed in LDS.
// ---------------------------------------------------------------------------
__launch_bounds__(128)
__global__ void flash_attn_wmma(const bf16* __restrict__ Q, const bf16* __restrict__ K,
                                const bf16* __restrict__ V, bf16* __restrict__ Oout) {
  __shared__ __align__(32) bf16 v_lds[64][32];     // [d][key]  (transposed V tile)
  __shared__ __align__(32) bf16 p_lds[4][16][32];  // [wave][qrow][key]

  const int tid  = threadIdx.x;
  const int wave = tid >> 5;
  const int lane = tid & 31;
  const int lo   = lane & 15;
  const int hi   = lane >> 4;

  const int qTile = blockIdx.x;   // 0..31
  const int h     = blockIdx.y;   // 0..15
  const int b     = blockIdx.z;   // 0..1

  const int rowBase = b * S_ + qTile * 64 + wave * 16;
  const int colBase = h * 64;

  // Q fragments, kept in registers for the whole kernel (2 x v16bf = 16 VGPRs)
  v16bf qf[2];
#pragma unroll
  for (int ks = 0; ks < 2; ++ks)
    qf[ks] = *reinterpret_cast<const v16bf*>(
        Q + (size_t)(rowBase + lo) * E_ + colBase + ks * 32 + hi * 16);

  v8f o[4] = {};                  // O accumulators: 16 q-rows x 64 d (fp32)
  float mrow[8], lrow[8];
#pragma unroll
  for (int r = 0; r < 8; ++r) { mrow[r] = -1e30f; lrow[r] = 0.0f; }

  for (int kt = 0; kt < S_; kt += 32) {
    __syncthreads();  // protect v_lds/p_lds reuse across iterations

    // Cooperative transpose-stage of V tile: V[kt..kt+32][colBase..+64] -> v_lds[d][key]
    {
      const int key = tid >> 2;           // 0..31
      const int d0  = (tid & 3) * 16;
      v16bf vv = *reinterpret_cast<const v16bf*>(
          V + (size_t)(b * S_ + kt + key) * E_ + colBase + d0);
#pragma unroll
      for (int j = 0; j < 16; ++j) v_lds[d0 + j][key] = vv[j];
    }

    // Scores: s[ns] = Q(16x64) @ K^T for keys [kt+ns*16, +16)
    v16bf kf[2][2];
#pragma unroll
    for (int ns = 0; ns < 2; ++ns)
#pragma unroll
      for (int ks = 0; ks < 2; ++ks)
        kf[ns][ks] = *reinterpret_cast<const v16bf*>(
            K + (size_t)(b * S_ + kt + ns * 16 + lo) * E_ + colBase + ks * 32 + hi * 16);

    v8f s[2] = {};
#pragma unroll
    for (int ns = 0; ns < 2; ++ns)
#pragma unroll
      for (int ks = 0; ks < 2; ++ks)
        s[ns] = WMMA_BF16(qf[ks], kf[ns][ks], s[ns]);

    // Online softmax. C-layout: VGPR r, lanes 0-15 => row r, lanes 16-31 => row r+8;
    // a row's 16 columns live across one 16-lane half -> xor-shuffle masks 1,2,4,8.
#pragma unroll
    for (int r = 0; r < 8; ++r) {
      float v0 = s[0][r] * SCALE_;
      float v1 = s[1][r] * SCALE_;
      float mx = fmaxf(v0, v1);
      mx = fmaxf(mx, __shfl_xor(mx, 1));
      mx = fmaxf(mx, __shfl_xor(mx, 2));
      mx = fmaxf(mx, __shfl_xor(mx, 4));
      mx = fmaxf(mx, __shfl_xor(mx, 8));
      float nm = fmaxf(mrow[r], mx);
      float p0 = __expf(v0 - nm);
      float p1 = __expf(v1 - nm);
      float rs = p0 + p1;
      rs += __shfl_xor(rs, 1);
      rs += __shfl_xor(rs, 2);
      rs += __shfl_xor(rs, 4);
      rs += __shfl_xor(rs, 8);
      float f = __expf(mrow[r] - nm);
      lrow[r] = lrow[r] * f + rs;
      mrow[r] = nm;
#pragma unroll
      for (int ds = 0; ds < 4; ++ds) o[ds][r] *= f;
      const int prow = r + 8 * hi;
      p_lds[wave][prow][lo]      = (bf16)p0;
      p_lds[wave][prow][16 + lo] = (bf16)p1;
    }

    __syncthreads();

    // O += P(16x32) @ V(32x64); both fragments are contiguous 32B LDS reads.
    v16bf pf = *reinterpret_cast<const v16bf*>(&p_lds[wave][lo][hi * 16]);
#pragma unroll
    for (int ds = 0; ds < 4; ++ds) {
      v16bf vf = *reinterpret_cast<const v16bf*>(&v_lds[ds * 16 + lo][hi * 16]);
      o[ds] = WMMA_BF16(pf, vf, o[ds]);
    }
  }

  // Normalize and store attention output (bf16 feed for the Wo GEMM)
#pragma unroll
  for (int r = 0; r < 8; ++r) {
    float inv = 1.0f / lrow[r];
    const int row = rowBase + r + 8 * hi;
#pragma unroll
    for (int ds = 0; ds < 4; ++ds)
      Oout[(size_t)row * E_ + colBase + ds * 16 + lo] = (bf16)(o[ds][r] * inv);
  }
}

// ---------------------------------------------------------------------------
// Row softmax over M=256 memory slots (applies SCALE), output bf16 weights.
// ---------------------------------------------------------------------------
__launch_bounds__(256)
__global__ void mem_softmax(const float* __restrict__ scores, bf16* __restrict__ w) {
  __shared__ float red[256];
  const int row = blockIdx.x;
  const int j   = threadIdx.x;
  float v = scores[(size_t)row * MEM_ + j] * SCALE_;
  red[j] = v;
  __syncthreads();
  for (int s = 128; s > 0; s >>= 1) {
    if (j < s) red[j] = fmaxf(red[j], red[j + s]);
    __syncthreads();
  }
  float mx = red[0];
  __syncthreads();
  float p = __expf(v - mx);
  red[j] = p;
  __syncthreads();
  for (int s = 128; s > 0; s >>= 1) {
    if (j < s) red[j] += red[j + s];
    __syncthreads();
  }
  w[(size_t)row * MEM_ + j] = (bf16)(p / red[0]);
}

// ---------------------------------------------------------------------------
// Host orchestration
// ---------------------------------------------------------------------------
extern "C" void kernel_launch(void* const* d_in, const int* in_sizes, int n_in,
                              void* d_out, int out_size, void* d_ws, size_t ws_size,
                              hipStream_t stream) {
  (void)in_sizes; (void)n_in; (void)out_size; (void)ws_size;
  const float* x   = (const float*)d_in[0];
  const float* Wq  = (const float*)d_in[1];
  const float* bq  = (const float*)d_in[2];
  const float* Wk  = (const float*)d_in[3];
  const float* bk  = (const float*)d_in[4];
  const float* Wv  = (const float*)d_in[5];
  const float* bv  = (const float*)d_in[6];
  const float* Wo  = (const float*)d_in[7];
  const float* bo  = (const float*)d_in[8];
  const float* mem = (const float*)d_in[9];

  char* ws = (char*)d_ws;
  size_t off = 0;
  auto alloc = [&](size_t bytes) -> void* {
    void* p = ws + off;
    off += (bytes + 255) & ~(size_t)255;
    return p;
  };

  bf16* x_bf    = (bf16*)alloc((size_t)ROWS_ * E_ * 2);
  bf16* q_bf    = (bf16*)alloc((size_t)ROWS_ * E_ * 2);
  bf16* k_bf    = (bf16*)alloc((size_t)ROWS_ * E_ * 2);
  bf16* v_bf    = (bf16*)alloc((size_t)ROWS_ * E_ * 2);
  bf16* wq_bf   = (bf16*)alloc((size_t)E_ * E_ * 2);
  bf16* wk_bf   = (bf16*)alloc((size_t)E_ * E_ * 2);
  bf16* wv_bf   = (bf16*)alloc((size_t)E_ * E_ * 2);
  bf16* wo_bf   = (bf16*)alloc((size_t)E_ * E_ * 2);
  bf16* mem_bf  = (bf16*)alloc((size_t)MEM_ * E_ * 2);
  bf16* memT_bf = (bf16*)alloc((size_t)E_ * MEM_ * 2);
  bf16* attn_bf = (bf16*)alloc((size_t)ROWS_ * E_ * 2);
  float* sc_f   = (float*)alloc((size_t)ROWS_ * MEM_ * 4);
  bf16* w_bf    = (bf16*)alloc((size_t)ROWS_ * MEM_ * 2);
  bf16* h_bf    = (bf16*)alloc((size_t)ROWS_ * E_ * 2);

  // 1) fp32 -> bf16 conversions
  int n;
  n = ROWS_ * E_; cvt_f32_bf16<<<(n + 255) / 256, 256, 0, stream>>>(x, x_bf, n);
  n = E_ * E_;    cvt_f32_bf16<<<(n + 255) / 256, 256, 0, stream>>>(Wq, wq_bf, n);
  n = E_ * E_;    cvt_f32_bf16<<<(n + 255) / 256, 256, 0, stream>>>(Wk, wk_bf, n);
  n = E_ * E_;    cvt_f32_bf16<<<(n + 255) / 256, 256, 0, stream>>>(Wv, wv_bf, n);
  n = E_ * E_;    cvt_f32_bf16<<<(n + 255) / 256, 256, 0, stream>>>(Wo, wo_bf, n);
  n = MEM_ * E_;  cvt_f32_bf16<<<(n + 255) / 256, 256, 0, stream>>>(mem, mem_bf, n);
  n = MEM_ * E_;  cvt_transpose_mem<<<(n + 255) / 256, 256, 0, stream>>>(mem, memT_bf);

  // 2) QKV projections: y = x @ W^T + b   (M=4096, N=1024, K=1024)
  dim3 gEE(E_ / 64, ROWS_ / 128);
  gemm_bf16_wmma<<<gEE, 128, 0, stream>>>(x_bf, wq_bf, bq, nullptr, 1.0f, q_bf, 1, E_, E_);
  gemm_bf16_wmma<<<gEE, 128, 0, stream>>>(x_bf, wk_bf, bk, nullptr, 1.0f, k_bf, 1, E_, E_);
  gemm_bf16_wmma<<<gEE, 128, 0, stream>>>(x_bf, wv_bf, bv, nullptr, 1.0f, v_bf, 1, E_, E_);

  // 3) Fused flash self-attention
  flash_attn_wmma<<<dim3(S_ / 64, H_, B_), 128, 0, stream>>>(q_bf, k_bf, v_bf, attn_bf);

  // 4) Memory cross-attention: scores = x @ memory^T  (M=4096, N=256, K=1024)
  gemm_bf16_wmma<<<dim3(MEM_ / 64, ROWS_ / 128), 128, 0, stream>>>(
      x_bf, mem_bf, nullptr, nullptr, 1.0f, sc_f, 0, MEM_, E_);
  mem_softmax<<<ROWS_, 256, 0, stream>>>(sc_f, w_bf);

  // 5) h = attn + 0.5 * (w @ memory)  (M=4096, N=1024, K=256; memory via memT)
  gemm_bf16_wmma<<<gEE, 128, 0, stream>>>(w_bf, memT_bf, nullptr, attn_bf, 0.5f, h_bf, 1, E_, MEM_);

  // 6) out = h @ Wo^T + bo  -> fp32 d_out
  gemm_bf16_wmma<<<gEE, 128, 0, stream>>>(h_bf, wo_bf, bo, nullptr, 1.0f, d_out, 0, E_, E_);
}
